// WindowAttention_51479478010538
// MI455X (gfx1250) — matrix operand order repaired and checked by reading, not compiled, not run on hardware
//
#include <hip/hip_runtime.h>
#include <hip/hip_bf16.h>

typedef unsigned short u16;
typedef float  v8f   __attribute__((ext_vector_type(8)));
typedef __bf16 v16bf __attribute__((ext_vector_type(16)));
typedef u16    u16x8 __attribute__((ext_vector_type(8)));
typedef u16    u16x4 __attribute__((ext_vector_type(4)));

#define DIM   384
#define NTOK  49
#define NPAD  64
#define NHEAD 12
#define HD    32
#define NWIN  4096
#define NW    64
#define SCALE 0.17677669529663689f

// LDS layout (bytes). Strides in bf16 elements (multiples of 8 -> 16B aligned rows).
#define XS 392              // xs / as row stride
#define QS 40               // qs / vs row stride
#define KT 72               // k-transposed row stride (m contiguous)
#define PS 72               // probs row stride
#define OFF_XS 0            // 64*392*2          = 50176   (dead after stage 2)
#define OFF_QS 50176        // 12*64*40*2        = 61440   (dead after S)
#define OFF_KT 111616       // 12*32*72*2        = 55296   (dead after S)
#define OFF_VS 166912       // 12*64*40*2        = 61440
#define OFF_PS 0            // 12*64*72*2        = 110592  (overlaps xs+qs, post-barrier)
#define OFF_AS 111616       // 64*392*2          = 50176   (overlaps kt, post-barrier)
#define SMEM_BYTES 228352

__device__ __forceinline__ u16 f2b(float f) {
    union { float f; unsigned u; } c; c.f = f;
    unsigned u = c.u;
    u += 0x7fffu + ((u >> 16) & 1u);          // round-to-nearest-even
    return (u16)(u >> 16);
}

__device__ __forceinline__ v16bf mkfrag(u16x8 lo, u16x8 hi) {
    union { struct { u16x8 a, b; } s; v16bf v; } u;
    u.s.a = lo; u.s.b = hi;
    return u.v;
}

// A-matrix 16x32 bf16 fragment (ISA 7.12.2): lane holds row M=lane&15,
// elements 0..7 -> K = k0 + 8*half + e, elements 8..15 -> K = k0 + 16 + 8*half + e.
template <typename P>
__device__ __forceinline__ v16bf ldA16(const P* base, int stride, int row, int k0, int hl) {
    const u16* p = (const u16*)base + row * stride + k0 + hl * 8;
    return mkfrag(*(const u16x8*)p, *(const u16x8*)(p + 16));
}

// B-matrix 32x16 bf16 fragment: lane holds column N = n0 + (lane&15),
// elements 0..7 -> K = kbase + 8*half + e, elements 8..15 -> K = kbase + 16 + 8*half + e.
template <typename P>
__device__ __forceinline__ v16bf ldB16(const P* base, int stride, int krow, int n0) {
    const u16* p = (const u16*)base + (krow & ~31) * stride + n0 + (krow & 15);
    const int hh = (krow >> 4) & 1;
    u16x8 lo, hi;
#pragma unroll
    for (int e = 0; e < 8; ++e) {
        lo[e] = p[(8 * hh + e) * stride];
        hi[e] = p[(16 + 8 * hh + e) * stride];
    }
    return mkfrag(lo, hi);
}

__device__ __forceinline__ v8f wmma_bf16(v16bf a, v16bf b, v8f c) {
    return __builtin_amdgcn_wmma_f32_16x16x32_bf16(false, a, false, b, (short)0, c, false, false);
}

// ---- prep kernels ---------------------------------------------------------
__global__ void prep_wqkv(const float* __restrict__ wq, u16* __restrict__ wt) {
    int idx = blockIdx.x * 256 + threadIdx.x;          // [K=384][N=1152]
    if (idx >= 384 * 1152) return;
    int k = idx / 1152, n = idx % 1152;
    wt[idx] = f2b(wq[n * 384 + k]);
}

__global__ void prep_wproj(const float* __restrict__ pw, u16* __restrict__ wt) {
    int idx = blockIdx.x * 256 + threadIdx.x;          // [K=384][N=384]
    if (idx >= 384 * 384) return;
    int k = idx / 384, n = idx % 384;
    wt[idx] = f2b(pw[n * 384 + k]);
}

// cb[w][h][m][mp] = rel_bias[h][m][mp] + mask[w][m][mp], padded to 64x64 with -1e30 key cols
__global__ void prep_cb(const float* __restrict__ mask, const float* __restrict__ tbl,
                        const int* __restrict__ rpi, float* __restrict__ cb) {
    int idx = blockIdx.x * 256 + threadIdx.x;
    if (idx >= NW * NHEAD * NPAD * NPAD) return;
    int mp = idx & 63;
    int m  = (idx >> 6) & 63;
    int h  = (idx >> 12) % NHEAD;
    int w  = idx / (NPAD * NPAD * NHEAD);
    float v;
    if (mp >= NTOK)      v = -1e30f;                    // padded key -> prob 0
    else if (m >= NTOK)  v = 0.0f;                      // padded query row (unused, keep finite)
    else v = tbl[rpi[m * NTOK + mp] * NHEAD + h] + mask[(w * NTOK + m) * NTOK + mp];
    cb[idx] = v;
}

// ---- fused per-window attention kernel ------------------------------------
__global__ void __launch_bounds__(384, 1)
winattn(const float* __restrict__ x, const float* __restrict__ qkv_b,
        const float* __restrict__ proj_b, const u16* __restrict__ wqkv,
        const u16* __restrict__ wproj, const float* __restrict__ cb,
        float* __restrict__ out) {
    __shared__ __align__(16) char smem[SMEM_BYTES];
    const int b    = blockIdx.x;
    const int tid  = threadIdx.x;
    const int w    = tid >> 5;          // wave 0..11
    const int lane = tid & 31;
    const int ln   = lane & 15;
    const int hl   = lane >> 4;

    u16* xs = (u16*)(smem + OFF_XS);
    const v8f zero = {0.f, 0.f, 0.f, 0.f, 0.f, 0.f, 0.f, 0.f};

    // ---- stage 1: x (49x384 f32) -> LDS bf16 (64x392, padded rows zero) ----
    {
        const float* xb = x + (size_t)b * NTOK * DIM;
        for (int c = tid; c < NPAD * (DIM / 4); c += 384) {
            int row = c / (DIM / 4), c4 = (c % (DIM / 4)) * 4;
            float4 f = make_float4(0.f, 0.f, 0.f, 0.f);
            if (row < NTOK) f = *(const float4*)(xb + row * DIM + c4);
            u16x4 p = { f2b(f.x), f2b(f.y), f2b(f.z), f2b(f.w) };
            *(u16x4*)(xs + row * XS + c4) = p;
        }
    }
    __syncthreads();

    // ---- stage 2: QKV = x @ Wqkv^T + b  (64x1152, K=384) -------------------
    // Wave owns 96 output columns (= one of q/k/v x 3 heads); sweeps all 4 M-tiles
    // with 4x2 register blocking so each weight fragment is fetched exactly once
    // per block (B-traffic = 1x Wqkv, L2-resident).
    {
        const int sec = w >> 2;                          // 0:q 1:k 2:v (wave-uniform)
        for (int pr = 0; pr < 3; ++pr) {                 // one head's 32 cols per pair
            const int n0 = w * 96 + pr * 32;             // global col base (j*16 offsets)
            v8f acc[4][2];
#pragma unroll
            for (int mt = 0; mt < 4; ++mt) { acc[mt][0] = zero; acc[mt][1] = zero; }

#pragma unroll
            for (int k = 0; k < 12; ++k) {
                v16bf B0 = ldB16(wqkv, 1152, k * 32 + lane, n0);
                v16bf B1 = ldB16(wqkv, 1152, k * 32 + lane, n0 + 16);
#pragma unroll
                for (int mt = 0; mt < 4; ++mt) {
                    v16bf A = ldA16(xs, XS, mt * 16 + ln, k * 32, hl);
                    acc[mt][0] = wmma_bf16(A, B0, acc[mt][0]);
                    acc[mt][1] = wmma_bf16(A, B1, acc[mt][1]);
                }
            }

            const int h2 = (w & 3) * 3 + pr;             // head within section
#pragma unroll
            for (int j = 0; j < 2; ++j) {
                float bias = qkv_b[n0 + j * 16 + ln];
                if (sec == 1) {                          // K: transposed [d][m], packed store
                    u16* kt = (u16*)(smem + OFF_KT) + h2 * (32 * KT);
                    int d = j * 16 + ln;
#pragma unroll
                    for (int mt = 0; mt < 4; ++mt) {
                        u16x8 pk;
#pragma unroll
                        for (int r = 0; r < 8; ++r) pk[r] = f2b(acc[mt][j][r] + bias);
                        *(u16x8*)(kt + d * KT + mt * 16 + hl * 8) = pk;
                    }
                } else {                                 // Q / V: row-major [m][d]
                    u16* dst = (u16*)(smem + (sec == 0 ? OFF_QS : OFF_VS)) + h2 * (NPAD * QS);
                    int col = j * 16 + ln;
#pragma unroll
                    for (int mt = 0; mt < 4; ++mt)
#pragma unroll
                        for (int r = 0; r < 8; ++r)
                            dst[(mt * 16 + hl * 8 + r) * QS + col] = f2b(acc[mt][j][r] + bias);
                }
            }
        }
    }
    __syncthreads();

    // ---- stage 3: per-head attention (wave == head) ------------------------
    {
        const int h = w;
        const u16* qh = (const u16*)(smem + OFF_QS) + h * (NPAD * QS);
        const u16* kh = (const u16*)(smem + OFF_KT) + h * (32 * KT);
        const u16* vh = (const u16*)(smem + OFF_VS) + h * (NPAD * QS);

        v16bf Qf[4], Kf[4];
#pragma unroll
        for (int mt = 0; mt < 4; ++mt) Qf[mt] = ldA16(qh, QS, mt * 16 + ln, 0, hl);
#pragma unroll
        for (int nt = 0; nt < 4; ++nt) Kf[nt] = ldB16(kh, KT, lane, nt * 16);

        const float* cbh = cb + ((size_t)(b & (NW - 1)) * NHEAD + h) * (NPAD * NPAD);
        v8f S[4][4];
#pragma unroll
        for (int mt = 0; mt < 4; ++mt)
#pragma unroll
            for (int nt = 0; nt < 4; ++nt) {
                v8f a = wmma_bf16(Qf[mt], Kf[nt], zero);
                int mp = nt * 16 + ln;
#pragma unroll
                for (int r = 0; r < 8; ++r)
                    a[r] = a[r] * SCALE + cbh[(mt * 16 + hl * 8 + r) * NPAD + mp];
                S[mt][nt] = a;
            }

        // row softmax: row lives in 16 lanes (same half) x 4 nt registers
#pragma unroll
        for (int mt = 0; mt < 4; ++mt)
#pragma unroll
            for (int r = 0; r < 8; ++r) {
                float mx = fmaxf(fmaxf(S[mt][0][r], S[mt][1][r]),
                                 fmaxf(S[mt][2][r], S[mt][3][r]));
#pragma unroll
                for (int o = 8; o >= 1; o >>= 1) mx = fmaxf(mx, __shfl_xor(mx, o, 32));
                float s = 0.f;
#pragma unroll
                for (int nt = 0; nt < 4; ++nt) {
                    float e = __expf(S[mt][nt][r] - mx);
                    S[mt][nt][r] = e; s += e;
                }
#pragma unroll
                for (int o = 8; o >= 1; o >>= 1) s += __shfl_xor(s, o, 32);
                float rs = 1.0f / s;
#pragma unroll
                for (int nt = 0; nt < 4; ++nt) S[mt][nt][r] *= rs;
            }

        __syncthreads();   // all waves done reading qs/ks/xs -> Ps/as regions reusable

        u16* Pp = (u16*)(smem + OFF_PS) + h * (NPAD * PS);
#pragma unroll
        for (int mt = 0; mt < 4; ++mt)
#pragma unroll
            for (int nt = 0; nt < 4; ++nt) {
                int mp = nt * 16 + ln;
#pragma unroll
                for (int r = 0; r < 8; ++r)
                    Pp[(mt * 16 + hl * 8 + r) * PS + mp] = f2b(S[mt][nt][r]);
            }

        v16bf Pf[4][2], Vf[2][2];
#pragma unroll
        for (int mt = 0; mt < 4; ++mt)
#pragma unroll
            for (int ks = 0; ks < 2; ++ks)
                Pf[mt][ks] = ldA16(Pp, PS, mt * 16 + ln, ks * 32, hl);
#pragma unroll
        for (int ks = 0; ks < 2; ++ks)
#pragma unroll
            for (int dt = 0; dt < 2; ++dt)
                Vf[ks][dt] = ldB16(vh, QS, lane + ks * 32, dt * 16);

        u16* asb = (u16*)(smem + OFF_AS);
#pragma unroll
        for (int mt = 0; mt < 4; ++mt)
#pragma unroll
            for (int dt = 0; dt < 2; ++dt) {
                v8f acc = zero;
#pragma unroll
                for (int ks = 0; ks < 2; ++ks) acc = wmma_bf16(Pf[mt][ks], Vf[ks][dt], acc);
                int col = h * HD + dt * 16 + ln;
#pragma unroll
                for (int r = 0; r < 8; ++r)
                    asb[(mt * 16 + hl * 8 + r) * XS + col] = f2b(acc[r]);
            }
    }
    __syncthreads();

    // ---- stage 4: out = attn_out @ Wproj^T + b  (64x384, K=384) ------------
    // Same 1x-weight-traffic scheme: wave owns 2 N-tiles, loops all 4 M-tiles.
    {
        const u16* asb = (const u16*)(smem + OFF_AS);
        const int n0 = w * 32;
        v8f acc[4][2];
#pragma unroll
        for (int mt = 0; mt < 4; ++mt) { acc[mt][0] = zero; acc[mt][1] = zero; }

#pragma unroll
        for (int k = 0; k < 12; ++k) {
            v16bf B0 = ldB16(wproj, 384, k * 32 + lane, n0);
            v16bf B1 = ldB16(wproj, 384, k * 32 + lane, n0 + 16);
#pragma unroll
            for (int mt = 0; mt < 4; ++mt) {
                v16bf A = ldA16(asb, XS, mt * 16 + ln, k * 32, hl);
                acc[mt][0] = wmma_bf16(A, B0, acc[mt][0]);
                acc[mt][1] = wmma_bf16(A, B1, acc[mt][1]);
            }
        }

#pragma unroll
        for (int j = 0; j < 2; ++j) {
            float pb = proj_b[n0 + j * 16 + ln];
#pragma unroll
            for (int mt = 0; mt < 4; ++mt)
#pragma unroll
                for (int r = 0; r < 8; ++r) {
                    int m = mt * 16 + hl * 8 + r;
                    if (m < NTOK)
                        out[((size_t)b * NTOK + m) * DIM + n0 + j * 16 + ln] = acc[mt][j][r] + pb;
                }
        }
    }
}

// ---- host entry -----------------------------------------------------------
extern "C" void kernel_launch(void* const* d_in, const int* in_sizes, int n_in,
                              void* d_out, int out_size, void* d_ws, size_t ws_size,
                              hipStream_t stream) {
    (void)in_sizes; (void)n_in; (void)out_size; (void)ws_size;
    const float* x      = (const float*)d_in[0];
    const float* mask   = (const float*)d_in[1];
    const float* qkv_w  = (const float*)d_in[2];
    const float* qkv_b  = (const float*)d_in[3];
    const float* proj_w = (const float*)d_in[4];
    const float* proj_b = (const float*)d_in[5];
    const float* tbl    = (const float*)d_in[6];
    const int*   rpi    = (const int*)d_in[7];
    float* out = (float*)d_out;

    char* ws = (char*)d_ws;
    u16*   wqkv  = (u16*)ws;                              //   884,736 B
    u16*   wproj = (u16*)(ws + 884736);                   //   294,912 B
    float* cb    = (float*)(ws + 884736 + 294912);        // 12,582,912 B

    prep_wqkv <<<(384 * 1152 + 255) / 256, 256, 0, stream>>>(qkv_w, wqkv);
    prep_wproj<<<(384 * 384 + 255) / 256, 256, 0, stream>>>(proj_w, wproj);
    prep_cb   <<<(NW * NHEAD * NPAD * NPAD + 255) / 256, 256, 0, stream>>>(mask, tbl, rpi, cb);
    winattn   <<<NWIN, 384, 0, stream>>>(x, qkv_b, proj_b, wqkv, wproj, cb, out);
}
